// LinearGaussianIndexed_22849226015349
// MI455X (gfx1250) — matrix operand order repaired and checked by reading, not compiled, not run on hardware
//
#include <hip/hip_runtime.h>
#include <hip/hip_bf16.h>

typedef __attribute__((ext_vector_type(16))) _Float16 v16h;
typedef __attribute__((ext_vector_type(2)))  _Float16 v2h;
typedef __attribute__((ext_vector_type(8)))  float    v8f;

#define NEXP 8
#define HD   64
#define IND  32
#define OD   16

// d_ws layout:
//   [0,256)                      : uint hdr[64]: cnt[8] @0, pad_off[8] @8, cursor[8] @16
//   [256, 256+128*512*2)         : packed f16 weight fragments (128 frags x 512 halves)
//   [256+131072, ...)            : int idxlist[N + 256]
#define WS_HDR_BYTES 256
#define WS_PACKW_BYTES (128 * 512 * 2)
#define WS_IDX_OFF (WS_HDR_BYTES + WS_PACKW_BYTES)

__global__ void k_zero(unsigned* hdr) {
  if (threadIdx.x < 64) hdr[threadIdx.x] = 0u;
}

// Pack all weights into WMMA B-fragment layout, f32 -> f16.
// Fragment r within an expert: r<4 = W1 n-tiles; 4..11 = W2 (n*2+kstep); 12..15 = W3 (n*2+kstep).
// B (32x16 f16) lane layout: lane<16: N=lane,   K=0..7, 16..23
//                            lane>=16: N=lane-16, K=8..15, 24..31
__global__ void k_pack(const float* __restrict__ W1, const float* __restrict__ W2,
                       const float* __restrict__ W3, _Float16* __restrict__ packW) {
  int f = blockIdx.x;          // 0..127
  int lane = threadIdx.x;      // 0..31
  int k = f >> 4, r = f & 15;
  int l15 = lane & 15;
  int koff = (lane < 16) ? 0 : 16;
  const float* src; int ld, n, k0;
  if (r < 4)       { src = W1 + (size_t)k * IND * HD;     ld = HD;     n = r;            k0 = 0; }
  else if (r < 12) { int t = r - 4;  src = W2 + (size_t)k * HD * HD;   ld = HD;     n = t >> 1; k0 = 32 * (t & 1); }
  else             { int t = r - 12; src = W3 + (size_t)k * HD * 2*OD; ld = 2 * OD; n = t >> 1; k0 = 32 * (t & 1); }
  _Float16* dst = packW + (size_t)f * 512 + (size_t)lane * 16;
  for (int h = 0; h < 16; ++h)
    dst[h] = (_Float16)src[(size_t)(k0 + (koff >> 1) + (h < 8 ? h : h + 8)) * ld + n * 16 + l15];
}

__global__ void k_count(const int* __restrict__ y, unsigned* hdr, int n) {
  int i = blockIdx.x * 256 + threadIdx.x;
  if (i < n) atomicAdd(&hdr[y[i] & 7], 1u);
}

__global__ void k_scan(unsigned* hdr) {
  unsigned off = 0;
  for (int k = 0; k < NEXP; ++k) {
    hdr[8 + k] = off;           // padded segment start (32-row granules)
    hdr[16 + k] = off;          // scatter cursor
    off += (hdr[k] + 31u) & ~31u;
  }
}

__global__ void k_scatter(const int* __restrict__ y, unsigned* hdr,
                          int* __restrict__ idxl, int n) {
  int i = blockIdx.x * 256 + threadIdx.x;
  if (i < n) {
    unsigned p = atomicAdd(&hdr[16 + (y[i] & 7)], 1u);
    idxl[p] = i;
  }
}

// One wave = two 16-row tiles (32 rows) of ONE expert segment (segments padded to 32).
// Weight fragments loaded once per wave, reused for both tiles: 32 WMMAs / wave.
__global__ __launch_bounds__(256) void
k_mlp(const float* __restrict__ x,
      const float* __restrict__ b1, const float* __restrict__ b2,
      const float* __restrict__ b3,
      const unsigned* __restrict__ hdr, const int* __restrict__ idxl,
      const v16h* __restrict__ packW, float* __restrict__ out, int n) {
  __shared__ __align__(32) _Float16 s_h[8][2][16 * HD];  // per-wave, per-tile 16x64 activations
  __shared__ __align__(16) int      s_idx[8][32];        // per-wave row indices (2 tiles)

  const int lane = threadIdx.x & 31;
  const int wave = threadIdx.x >> 5;
  const int l15  = lane & 15;
  const int c0   = (lane < 16) ? 0 : 8;   // A-fragment K sub-offset
  const int moff = (lane < 16) ? 0 : 8;   // C/D M sub-offset
  const long p = ((long)blockIdx.x * 8 + wave) * 32;

  // Lane-parallel expert-segment lookup: lane k (k<8) tests expert k.
  unsigned c_me  = hdr[lane & 7];
  unsigned po_me = hdr[8 + (lane & 7)];
  long pc_me = (long)((c_me + 31u) & ~31u);
  unsigned inm = (unsigned)__ballot((lane < 8) && p >= (long)po_me &&
                                    p < (long)po_me + pc_me);
  if (inm == 0u) return;  // past total padded rows (no block barriers -> safe)
  const int seg = __builtin_ctz(inm);
  const unsigned scnt  = (unsigned)__shfl((int)c_me, seg, 32);
  const unsigned po_sg = (unsigned)__shfl((int)po_me, seg, 32);
  const long rowsb = p - (long)po_sg;

  // Row indices + validity for all 32 rows (lane == row).
  {
    int vr = (unsigned)(rowsb + lane) < scnt;
    s_idx[wave][lane] = vr ? idxl[p + lane] : 0;
  }
  unsigned vm_all = (unsigned)__ballot((unsigned)(rowsb + lane) < scnt);
  asm volatile("" ::: "memory");

  const int M = l15;

  // A0 per tile: x tile (16x32), f32 -> f16 (halves 0..7 = K c0.., 8..15 = K c0+16..)
  v16h a0[2];
  for (int t = 0; t < 2; ++t) {
    const int myidx = s_idx[wave][t * 16 + M];
    const float* xr = x + (size_t)myidx * IND;
    float4 f0 = *(const float4*)(xr + c0);
    float4 f1 = *(const float4*)(xr + c0 + 4);
    float4 f2 = *(const float4*)(xr + c0 + 16);
    float4 f3 = *(const float4*)(xr + c0 + 20);
    v16h a;
    a[0]=(_Float16)f0.x; a[1]=(_Float16)f0.y; a[2]=(_Float16)f0.z; a[3]=(_Float16)f0.w;
    a[4]=(_Float16)f1.x; a[5]=(_Float16)f1.y; a[6]=(_Float16)f1.z; a[7]=(_Float16)f1.w;
    a[8]=(_Float16)f2.x; a[9]=(_Float16)f2.y; a[10]=(_Float16)f2.z; a[11]=(_Float16)f2.w;
    a[12]=(_Float16)f3.x; a[13]=(_Float16)f3.y; a[14]=(_Float16)f3.z; a[15]=(_Float16)f3.w;
    a0[t] = a;
  }

  auto frag = [&](int r) -> v16h {
    return packW[((size_t)seg * 16 + r) * 32 + lane];
  };
  union UV { v16h v; uint4 q[2]; };
  auto ldA = [&](int t, int K0) -> v16h {  // 16x32 A-fragment from s_h[t][M][64]
    UV u;
    u.q[0] = *(const uint4*)&s_h[wave][t][M * HD + K0 + c0];
    u.q[1] = *(const uint4*)&s_h[wave][t][M * HD + K0 + c0 + 16];
    return u.v;
  };
  // Packed f16 convert (v_cvt_pkrtz_f16_f32) + pair of b16 stores (lo + d16_hi).
  auto stAcc = [&](int t, const v8f& c, int nn) {
    for (int m = 0; m < 8; m += 2) {
      v2h pk = __builtin_bit_cast(v2h, __builtin_amdgcn_cvt_pkrtz(c[m], c[m + 1]));
      s_h[wave][t][(m + moff) * HD + nn * 16 + l15]     = pk[0];
      s_h[wave][t][(m + 1 + moff) * HD + nn * 16 + l15] = pk[1];
    }
  };

  // Per-lane bias values (N = l15 within each n-tile), shared by both tiles.
  float bv1[4], bv2[4], bv3[2];
  for (int nn = 0; nn < 4; ++nn) bv1[nn] = b1[seg * HD + nn * 16 + l15];
  for (int nn = 0; nn < 4; ++nn) bv2[nn] = b2[seg * HD + nn * 16 + l15];
  for (int nn = 0; nn < 2; ++nn) bv3[nn] = b3[seg * 2 * OD + nn * 16 + l15];

  const v8f cz = {};  // lowers to WMMA SRC2 = inline 0

  // ---- Layer 1: (16x32) @ (32x64), then fused +bias, ReLU ----
  v8f h[2][4];
  for (int nn = 0; nn < 4; ++nn) {
    v16h b = frag(nn);
    for (int t = 0; t < 2; ++t) {
      v8f c = __builtin_amdgcn_wmma_f32_16x16x32_f16(false, a0[t], false, b,
                                                     (short)0, cz, false, false);
      for (int i = 0; i < 8; ++i) c[i] = fmaxf(c[i] + bv1[nn], 0.f);
      h[t][nn] = c;
    }
  }
  asm volatile("" ::: "memory");
  for (int t = 0; t < 2; ++t)
    for (int nn = 0; nn < 4; ++nn)
      stAcc(t, h[t][nn], nn);
  asm volatile("" ::: "memory");

  // ---- Layer 2: (16x64) @ (64x64), fused +bias, ReLU ----
  v16h a2[2][2];
  for (int t = 0; t < 2; ++t) { a2[t][0] = ldA(t, 0); a2[t][1] = ldA(t, 32); }
  for (int nn = 0; nn < 4; ++nn) {
    v16h bA = frag(4 + nn * 2), bB = frag(5 + nn * 2);
    for (int t = 0; t < 2; ++t) {
      v8f c = __builtin_amdgcn_wmma_f32_16x16x32_f16(false, a2[t][0], false, bA,
                                                     (short)0, cz, false, false);
      c = __builtin_amdgcn_wmma_f32_16x16x32_f16(false, a2[t][1], false, bB,
                                                 (short)0, c, false, false);
      for (int i = 0; i < 8; ++i) c[i] = fmaxf(c[i] + bv2[nn], 0.f);
      h[t][nn] = c;
    }
  }
  asm volatile("" ::: "memory");
  for (int t = 0; t < 2; ++t)
    for (int nn = 0; nn < 4; ++nn)
      stAcc(t, h[t][nn], nn);
  asm volatile("" ::: "memory");

  // ---- Layer 3: (16x64) @ (64x32); bias added in epilogue ----
  v16h a3[2][2];
  for (int t = 0; t < 2; ++t) { a3[t][0] = ldA(t, 0); a3[t][1] = ldA(t, 32); }
  v8f o[2][2];
  for (int nn = 0; nn < 2; ++nn) {
    v16h bA = frag(12 + nn * 2), bB = frag(13 + nn * 2);
    for (int t = 0; t < 2; ++t) {
      v8f c = __builtin_amdgcn_wmma_f32_16x16x32_f16(false, a3[t][0], false, bA,
                                                     (short)0, cz, false, false);
      c = __builtin_amdgcn_wmma_f32_16x16x32_f16(false, a3[t][1], false, bB,
                                                 (short)0, c, false, false);
      o[t][nn] = c;
    }
  }

  // Scatter outputs: mu = o[:, :16] + b3, sigma = softplus(o[:, 16:] + b3)
  const size_t sigoff = (size_t)n * OD;
  for (int t = 0; t < 2; ++t) {
    int4 ia = *(const int4*)&s_idx[wave][t * 16 + moff];
    int4 ib = *(const int4*)&s_idx[wave][t * 16 + moff + 4];
    int rows[8] = {ia.x, ia.y, ia.z, ia.w, ib.x, ib.y, ib.z, ib.w};
    unsigned vmask = vm_all >> (t * 16);
    for (int m = 0; m < 8; ++m) {
      int Mr = m + moff;
      if ((vmask >> Mr) & 1u) {
        size_t row = (size_t)rows[m];
        out[row * OD + l15] = o[t][0][m] + bv3[0];
        float v = o[t][1][m] + bv3[1];
        float sp = fmaxf(v, 0.f) + __logf(1.f + __expf(-fabsf(v)));
        out[sigoff + row * OD + l15] = sp;
      }
    }
  }
}

extern "C" void kernel_launch(void* const* d_in, const int* in_sizes, int n_in,
                              void* d_out, int out_size, void* d_ws, size_t ws_size,
                              hipStream_t stream) {
  const float* x  = (const float*)d_in[0];
  const int*   y  = (const int*)d_in[1];
  const float* W1 = (const float*)d_in[2];
  const float* b1 = (const float*)d_in[3];
  const float* W2 = (const float*)d_in[4];
  const float* b2 = (const float*)d_in[5];
  const float* W3 = (const float*)d_in[6];
  const float* b3 = (const float*)d_in[7];
  float* out = (float*)d_out;
  const int n = in_sizes[0] / IND;

  unsigned*  hdr   = (unsigned*)d_ws;
  _Float16*  packW = (_Float16*)((char*)d_ws + WS_HDR_BYTES);
  int*       idxl  = (int*)((char*)d_ws + WS_IDX_OFF);

  k_zero<<<1, 64, 0, stream>>>(hdr);
  k_pack<<<128, 32, 0, stream>>>(W1, W2, W3, packW);
  const int nb = (n + 255) / 256;
  k_count<<<nb, 256, 0, stream>>>(y, hdr, n);
  k_scan<<<1, 1, 0, stream>>>(hdr);
  k_scatter<<<nb, 256, 0, stream>>>(y, hdr, idxl, n);
  const int gb = (n + NEXP * 31 + 255) / 256;  // covers all 32-row-padded segments
  k_mlp<<<gb, 256, 0, stream>>>(x, b1, b2, b3, hdr, idxl,
                                (const v16h*)packW, out, n);
}
